// ContrastiveLoss_2911987827481
// MI455X (gfx1250) — matrix operand order repaired and checked by reading, not compiled, not run on hardware
//
#include <hip/hip_runtime.h>
#include <hip/hip_bf16.h>
#include <math.h>

// ---------------------------------------------------------------------------
// NT-Xent contrastive loss, fused for MI455X (gfx1250, wave32, WMMA bf16)
//   emb_i, emb_j : [4096, 1024] fp32 -> scalar fp32 loss
// sim = reps@reps^T fused with masked exp row-sums; sim (256 MB) never
// materialized. bf16 WMMA / fp32 accum (logits in [-2,2], no LSE needed).
// sim is SYMMETRIC: only the upper-triangular 256x256 super-tiles are
// evaluated (528 of 1024); strict-upper tiles scatter exp sums to both
// denom[row] (row reduction) and denom[col] (mirror, column reduction).
// K-slabs DMA'd global->LDS with GLOBAL_LOAD_ASYNC_TO_LDS_B128 (ASYNCcnt)
// into a double buffer overlapping the 16-WMMA block; zero staging VGPRs.
// ---------------------------------------------------------------------------

typedef __attribute__((ext_vector_type(16))) __bf16 v16bf;
typedef __attribute__((ext_vector_type(8)))  __bf16 v8bf;
typedef __attribute__((ext_vector_type(8)))  float  v8f;

#define N_ROWS 4096
#define DIM    1024
#define TWO_N  8192
#define INV_T  2.0f      // 1 / TEMPERATURE
#define EPS_N  1e-12f
#define NSUP   32        // 8192 / 256 super-tiles per side
#define NPAIR  ((NSUP * (NSUP + 1)) / 2)   // 528 upper-tri super-tiles

// ---------------------------------------------------------------------------
// Kernel 0: zero the accumulators (denom[8192], out[1])
// ---------------------------------------------------------------------------
__global__ void cl_init_kernel(float* __restrict__ denom, float* __restrict__ out) {
    int i = blockIdx.x * blockDim.x + threadIdx.x;
    if (i < TWO_N) denom[i] = 0.0f;
    if (i == 0) out[0] = 0.0f;
}

// ---------------------------------------------------------------------------
// Kernel 1: L2-normalize each row (fp32 math) and store as bf16.
// ---------------------------------------------------------------------------
__global__ __launch_bounds__(256)
void cl_norm_kernel(const float* __restrict__ emb_i,
                    const float* __restrict__ emb_j,
                    __bf16* __restrict__ reps) {
    const int row = blockIdx.x;                       // 0..8191
    const float* src = (row < N_ROWS)
        ? (emb_i + (size_t)row * DIM)
        : (emb_j + (size_t)(row - N_ROWS) * DIM);
    const int t = threadIdx.x;

    float4 x = ((const float4*)src)[t];               // 256*4 = 1024 elements
    float ss = x.x * x.x + x.y * x.y + x.z * x.z + x.w * x.w;

    for (int m = 16; m >= 1; m >>= 1) ss += __shfl_xor(ss, m, 32);

    __shared__ float s_part[8];
    __shared__ float s_scale;
    const int lane = t & 31, w = t >> 5;
    if (lane == 0) s_part[w] = ss;
    __syncthreads();
    if (t == 0) {
        float s = 0.0f;
        for (int i = 0; i < 8; ++i) s += s_part[i];
        s_scale = 1.0f / fmaxf(sqrtf(s), EPS_N);
    }
    __syncthreads();
    const float sc = s_scale;

    __bf16* dst = reps + (size_t)row * DIM + t * 4;
    dst[0] = (__bf16)(x.x * sc);
    dst[1] = (__bf16)(x.y * sc);
    dst[2] = (__bf16)(x.z * sc);
    dst[3] = (__bf16)(x.w * sc);
}

// ---------------------------------------------------------------------------
// Kernel 2: positives[k] = dot(z_i[k], z_j[k])  (bf16 in, fp32 accum)
// ---------------------------------------------------------------------------
__global__ __launch_bounds__(256)
void cl_pos_kernel(const __bf16* __restrict__ reps, float* __restrict__ pos) {
    const int k = blockIdx.x;                         // 0..4095
    const int t = threadIdx.x;
    const __bf16* a = reps + (size_t)k * DIM;
    const __bf16* b = reps + (size_t)(k + N_ROWS) * DIM;

    float s = 0.0f;
    #pragma unroll
    for (int i = 0; i < 4; ++i) {
        int idx = t * 4 + i;
        s += (float)a[idx] * (float)b[idx];
    }
    for (int m = 16; m >= 1; m >>= 1) s += __shfl_xor(s, m, 32);

    __shared__ float s_part[8];
    const int lane = t & 31, w = t >> 5;
    if (lane == 0) s_part[w] = s;
    __syncthreads();
    if (t == 0) {
        float tot = 0.0f;
        for (int i = 0; i < 8; ++i) tot += s_part[i];
        pos[k] = tot;
    }
}

// ---------------------------------------------------------------------------
// Kernel 3: fused triangular sim GEMM + exp row/col scatter.
// 1D grid of 2*NPAIR blocks; block p>>1 = upper-tri super-tile (i<=j),
// p&1 = which 128-row half. Block tile 128(M) x 256(N); 8 waves 2x4;
// wave tile 64x64 = 4x4 WMMA 16x16x32 bf16.
// ---------------------------------------------------------------------------
__global__ __launch_bounds__(256)
void cl_simdenom_kernel(const __bf16* __restrict__ reps,
                        float* __restrict__ denom) {
    __shared__ __align__(32) __bf16 sA[2][128 * 32];  // 2 x 8 KB
    __shared__ __align__(32) __bf16 sB[2][256 * 32];  // 2 x 16 KB

    // ---- decode triangular super-tile (i <= j) ----
    const int p    = blockIdx.x >> 1;     // 0..527
    const int half = blockIdx.x & 1;
    auto tri_base = [](int a) { return a * NSUP - (a * (a - 1)) / 2; };
    int i = (int)((65.0f - sqrtf(4225.0f - 8.0f * (float)p)) * 0.5f);
    if (i < 0) i = 0;
    if (i > NSUP - 1) i = NSUP - 1;
    while (i + 1 <= NSUP - 1 && tri_base(i + 1) <= p) ++i;
    while (i > 0 && tri_base(i) > p) --i;
    const int j = i + (p - tri_base(i));
    const bool mirror = (i != j);         // strict-upper: scatter transpose too

    const int rowBase = i * 256 + half * 128;
    const int colBase = j * 256;

    const int t    = threadIdx.x;
    const int lane = t & 31;
    const int w    = t >> 5;          // wave 0..7
    const int wm   = w >> 2;          // 0..1  -> 64-row strip
    const int wn   = w & 3;           // 0..3  -> 64-col strip

    v8f acc[4][4] = {};

    const int halfSel = lane >> 4;    // 0: lanes 0-15, 1: lanes 16-31
    const int lrow    = lane & 15;

    // Async global->LDS DMA of one 32-wide K-slab into buffer `buf`.
    // 16B chunks: A slab = 512 chunks (2/thread), B slab = 1024 chunks (4/thread).
    auto issue_async = [&](int k0, int buf) {
        #pragma unroll
        for (int q = 0; q < 2; ++q) {
            int c = t + 256 * q, r = c >> 2, col = (c & 3) * 8;
            unsigned lds = (unsigned)(size_t)&sA[buf][r * 32 + col];   // low 32 bits of pointer = LDS address
            unsigned long long g =
                (unsigned long long)(size_t)(reps + (size_t)(rowBase + r) * DIM + k0 + col);
            asm volatile("global_load_async_to_lds_b128 %0, %1, off"
                         :: "v"(lds), "v"(g) : "memory");
        }
        #pragma unroll
        for (int q = 0; q < 4; ++q) {
            int c = t + 256 * q, r = c >> 2, col = (c & 3) * 8;
            unsigned lds = (unsigned)(size_t)&sB[buf][r * 32 + col];
            unsigned long long g =
                (unsigned long long)(size_t)(reps + (size_t)(colBase + r) * DIM + k0 + col);
            asm volatile("global_load_async_to_lds_b128 %0, %1, off"
                         :: "v"(lds), "v"(g) : "memory");
        }
    };

    issue_async(0, 0);
    asm volatile("s_wait_asynccnt 0x0" ::: "memory");
    __syncthreads();

    for (int k0 = 0; k0 < DIM; k0 += 32) {
        const int  buf  = (k0 >> 5) & 1;
        const bool more = (k0 + 32) < DIM;
        if (more) issue_async(k0 + 32, buf ^ 1);   // DMA in flight during WMMAs

        // B fragments resident (4 x v16bf); A fragment streamed per mt.
        v16bf bfrag[4];
        #pragma unroll
        for (int nt = 0; nt < 4; ++nt) {
            // B 32x16 bf16: lane = column, K contiguous 16 per half-wave
            const __bf16* base = &sB[buf][(wn * 64 + nt * 16 + lrow) * 32];
            bfrag[nt] = *(const v16bf*)(base + halfSel * 16);
        }
        #pragma unroll
        for (int mt = 0; mt < 4; ++mt) {
            // A 16x32 bf16: lanes 0-15 hold K{0..15}, lanes 16-31 K{16..31}
            const __bf16* base = &sA[buf][(wm * 64 + mt * 16 + lrow) * 32];
            v16bf afrag = *(const v16bf*)(base + halfSel * 16);
            #pragma unroll
            for (int nt = 0; nt < 4; ++nt)
                acc[mt][nt] = __builtin_amdgcn_wmma_f32_16x16x32_bf16(
                    false, afrag, false, bfrag[nt],
                    (short)0, acc[mt][nt], false, false);
        }

        if (more) asm volatile("s_wait_asynccnt 0x0" ::: "memory");
        __syncthreads();   // readers of buf done; buf^1 fully written
    }

    // ---- fused masked exp + row/col scatter ----
    // C/D layout: VGPR r, lanes 0-15 -> M=r, lanes 16-31 -> M=r+8; N = lane&15.
    // Row sums: accumulate exp across nt, one 16-lane reduction + atomic per
    // (mt,r). Column sums (mirror): a lane's entries per nt tile share one
    // column, so total = local sum + shfl_xor(...,16) partner half.
    const int lcol = lane & 15;
    float colsum[4] = {0.0f, 0.0f, 0.0f, 0.0f};
    #pragma unroll
    for (int mt = 0; mt < 4; ++mt) {
        #pragma unroll
        for (int r = 0; r < 8; ++r) {
            const int row = rowBase + wm * 64 + mt * 16 + r + halfSel * 8;
            float rs = 0.0f;
            #pragma unroll
            for (int nt = 0; nt < 4; ++nt) {
                const int col = colBase + wn * 64 + nt * 16 + lcol;
                const float ev = (row == col) ? 0.0f : __expf(acc[mt][nt][r] * INV_T);
                rs += ev;
                colsum[nt] += ev;
            }
            rs += __shfl_xor(rs, 1, 32);
            rs += __shfl_xor(rs, 2, 32);
            rs += __shfl_xor(rs, 4, 32);
            rs += __shfl_xor(rs, 8, 32);
            if (lcol == 0) atomicAdd(&denom[row], rs);
        }
    }
    if (mirror) {
        #pragma unroll
        for (int nt = 0; nt < 4; ++nt) {
            const float cs = colsum[nt] + __shfl_xor(colsum[nt], 16, 32);
            if (lane < 16)
                atomicAdd(&denom[colBase + wn * 64 + nt * 16 + lcol], cs);
        }
    }
}

// ---------------------------------------------------------------------------
// Kernel 4: loss = mean_k -(pos/T - log(denom_k))
// ---------------------------------------------------------------------------
__global__ __launch_bounds__(256)
void cl_finalize_kernel(const float* __restrict__ pos,
                        const float* __restrict__ denom,
                        float* __restrict__ out) {
    const int k = blockIdx.x * blockDim.x + threadIdx.x;  // 32 blocks * 256 = 8192
    float v = 0.0f;
    if (k < TWO_N) {
        const float p = pos[(k < N_ROWS) ? k : (k - N_ROWS)];
        v = -(p * INV_T - __logf(denom[k]));
    }
    for (int m = 16; m >= 1; m >>= 1) v += __shfl_xor(v, m, 32);

    __shared__ float s_part[8];
    const int lane = threadIdx.x & 31, w = threadIdx.x >> 5;
    if (lane == 0) s_part[w] = v;
    __syncthreads();
    if (threadIdx.x == 0) {
        float tot = 0.0f;
        for (int i = 0; i < 8; ++i) tot += s_part[i];
        atomicAdd(out, tot * (1.0f / (float)TWO_N));
    }
}

// ---------------------------------------------------------------------------
extern "C" void kernel_launch(void* const* d_in, const int* in_sizes, int n_in,
                              void* d_out, int out_size, void* d_ws, size_t ws_size,
                              hipStream_t stream) {
    const float* emb_i = (const float*)d_in[0];
    const float* emb_j = (const float*)d_in[1];
    float* out = (float*)d_out;

    // workspace layout
    char* ws = (char*)d_ws;
    __bf16* reps  = (__bf16*)ws;                                   // 8192*1024*2 = 16 MB
    float*  denom = (float*)(ws + (size_t)TWO_N * DIM * 2);        // 8192 * 4
    float*  pos   = (float*)(ws + (size_t)TWO_N * DIM * 2
                                + (size_t)TWO_N * sizeof(float));  // 4096 * 4

    cl_init_kernel<<<(TWO_N + 255) / 256, 256, 0, stream>>>(denom, out);
    cl_norm_kernel<<<TWO_N, 256, 0, stream>>>(emb_i, emb_j, reps);
    cl_pos_kernel<<<N_ROWS, 256, 0, stream>>>(reps, pos);

    cl_simdenom_kernel<<<2 * NPAIR, 256, 0, stream>>>(reps, denom);  // 1056 blocks

    cl_finalize_kernel<<<TWO_N / 256, 256, 0, stream>>>(pos, denom, out);
}
